// GraphAttentionLayer_16054587752703
// MI455X (gfx1250) — matrix operand order repaired and checked by reading, not compiled, not run on hardware
//
#include <hip/hip_runtime.h>

// ---------------------------------------------------------------------------
// GAT layer for MI455X (gfx1250, wave32, WMMA).
// B=16, N=512, E=256, H=8, D=32, alpha=0.2
// Inputs (f32): x[B,N,E], A[B,N,N], W[E,E], bW[E], a[E,2H], ba[2H]
// Output (f32): [B,N,E]
// ---------------------------------------------------------------------------

typedef __bf16 bf16_t;
typedef __bf16 v16bf __attribute__((ext_vector_type(16)));
typedef __bf16 v8bf  __attribute__((ext_vector_type(8)));
typedef float  v8f   __attribute__((ext_vector_type(8)));

constexpr int BB = 16;
constexpr int NN = 512;
constexpr int EE = 256;
constexpr int HH = 8;
constexpr int DD = 32;   // E/H
constexpr float ALPHA = 0.2f;

// ---------------------------------------------------------------------------
// Kernel 0: convert x -> bf16 (row major) and W -> bf16 transposed Wbt[f][e]
// ---------------------------------------------------------------------------
__global__ __launch_bounds__(256) void gat_prep(
    const float* __restrict__ x, const float* __restrict__ W,
    bf16_t* __restrict__ xb, bf16_t* __restrict__ Wbt)
{
  const int idx = blockIdx.x * 256 + threadIdx.x;
  constexpr int nx = BB * NN * EE;
  if (idx < nx) {
    xb[idx] = (bf16_t)x[idx];
  } else {
    const int j = idx - nx;          // j = f*EE + e
    if (j < EE * EE) {
      const int f = j >> 8;
      const int e = j & 255;
      Wbt[j] = (bf16_t)W[e * EE + f];
    }
  }
}

// ---------------------------------------------------------------------------
// Kernel 1: Wh = x @ W + bW, stored bf16 TRANSPOSED: WhT[b][f][n]
// One block = one 16-row tile; 8 waves x (2 x 16-col tiles) = 256 cols.
// K loop: 8 steps of 32 (v_wmma_f32_16x16x32_bf16).
// ---------------------------------------------------------------------------
__global__ __launch_bounds__(256) void gat_gemm1(
    const bf16_t* __restrict__ xb, const bf16_t* __restrict__ Wbt,
    const float* __restrict__ bW, bf16_t* __restrict__ WhT)
{
  const int r0  = blockIdx.x * 16;        // global row (b*N + n)
  const int b   = r0 >> 9;
  const int n0  = r0 & (NN - 1);
  const int w   = threadIdx.x >> 5;       // wave id -> 32-col group
  const int L   = threadIdx.x & 31;
  const int l16 = L & 15;
  const int hi  = L >> 4;                 // lane half selects K sub-block

  v8f acc0 = {};
  v8f acc1 = {};
  const bf16_t* arow = xb + (size_t)(r0 + l16) * EE;

  #pragma unroll
  for (int kk = 0; kk < 8; ++kk) {
    const int k0 = kk * 32;
    // A fragment: row l16, K = {hi*8..hi*8+7, 16+hi*8..}
    v8bf alo = *reinterpret_cast<const v8bf*>(arow + k0 + hi * 8);
    v8bf ahi = *reinterpret_cast<const v8bf*>(arow + k0 + 16 + hi * 8);
    v16bf amat;
    #pragma unroll
    for (int i = 0; i < 8; ++i) { amat[i] = alo[i]; amat[8 + i] = ahi[i]; }

    #pragma unroll
    for (int nt = 0; nt < 2; ++nt) {
      const int col = w * 32 + nt * 16 + l16;     // N index of this lane
      const bf16_t* bp = Wbt + (size_t)col * EE + k0 + hi * 8;
      v8bf b0 = *reinterpret_cast<const v8bf*>(bp);
      v8bf b1 = *reinterpret_cast<const v8bf*>(bp + 16);
      v16bf bmat;
      #pragma unroll
      for (int i = 0; i < 8; ++i) { bmat[i] = b0[i]; bmat[8 + i] = b1[i]; }
      if (nt == 0)
        acc0 = __builtin_amdgcn_wmma_f32_16x16x32_bf16(
            false, amat, false, bmat, (short)0, acc0, false, false);
      else
        acc1 = __builtin_amdgcn_wmma_f32_16x16x32_bf16(
            false, amat, false, bmat, (short)0, acc1, false, false);
    }
  }

  // Store transposed (+bias): C vgpr g -> row n0 + hi*8 + g, col = lane N
  #pragma unroll
  for (int nt = 0; nt < 2; ++nt) {
    const int col = w * 32 + nt * 16 + l16;
    const float bias = bW[col];
    const v8f acc = nt ? acc1 : acc0;
    v8bf o;
    #pragma unroll
    for (int g = 0; g < 8; ++g) o[g] = (bf16_t)(acc[g] + bias);
    *reinterpret_cast<v8bf*>(WhT + (size_t)(b * EE + col) * NN + n0 + hi * 8) = o;
  }
}

// ---------------------------------------------------------------------------
// Kernel 2: attention logits  E1[b][h][n] = Wh[b,n,:]·a[:,2h]   + ba[2h]
//                             E2[b][h][n] = Wh[b,n,:]·a[:,2h+1] + ba[2h+1]
// ---------------------------------------------------------------------------
__global__ __launch_bounds__(256) void gat_e12(
    const bf16_t* __restrict__ WhT, const float* __restrict__ a,
    const float* __restrict__ ba, float* __restrict__ E1,
    float* __restrict__ E2)
{
  const int tid = blockIdx.x * 256 + threadIdx.x;   // [b][c][n]
  const int n = tid & (NN - 1);
  const int c = (tid >> 9) & 15;                    // c = 2h + which
  const int b = tid >> 13;
  float acc = ba[c];
  const bf16_t* colp = WhT + (size_t)b * EE * NN + n;
  #pragma unroll 4
  for (int e = 0; e < EE; ++e)
    acc += (float)colp[(size_t)e * NN] * a[e * (2 * HH) + c];
  const int h = c >> 1;
  float* dst = (c & 1) ? E2 : E1;
  dst[(size_t)(b * HH + h) * NN + n] = acc;
}

// ---------------------------------------------------------------------------
// Kernel 3: masked leaky-relu softmax + att @ Whh + ReLU.
// Block = one (b, 16-row tile); wave w = head h.
// A-mask tile (16x512 f32, 32KB) + per-head E2 (8x512 f32, 16KB) staged in
// LDS once and shared by all 8 heads across both passes (CDNA5: 320KB/WGP).
// Per-lane j assignment mirrors the WMMA A-operand layout exactly.
// ---------------------------------------------------------------------------
__device__ inline void ld16(const float* __restrict__ p, float v[16]) {
  float4 t0 = *reinterpret_cast<const float4*>(p);
  float4 t1 = *reinterpret_cast<const float4*>(p + 4);
  float4 t2 = *reinterpret_cast<const float4*>(p + 16);
  float4 t3 = *reinterpret_cast<const float4*>(p + 20);
  v[0]=t0.x; v[1]=t0.y; v[2]=t0.z; v[3]=t0.w;
  v[4]=t1.x; v[5]=t1.y; v[6]=t1.z; v[7]=t1.w;
  v[8]=t2.x; v[9]=t2.y; v[10]=t2.z; v[11]=t2.w;
  v[12]=t3.x; v[13]=t3.y; v[14]=t3.z; v[15]=t3.w;
}

__global__ __launch_bounds__(256) void gat_attn(
    const float* __restrict__ Adj, const float* __restrict__ E1,
    const float* __restrict__ E2, const bf16_t* __restrict__ WhT,
    float* __restrict__ out)
{
  __shared__ float Als[16 * NN];    // A-mask tile, rows i0..i0+15
  __shared__ float E2ls[HH * NN];   // e2 per head

  const int b   = blockIdx.x >> 5;
  const int i0  = (blockIdx.x & 31) * 16;
  const int h   = threadIdx.x >> 5;       // head
  const int L   = threadIdx.x & 31;
  const int l16 = L & 15;
  const int hi  = L >> 4;
  const int base = hi * 8;                // j sub-offset for this lane half
  const int koff = hi * 8;                // B operand K offset

  // ---- cooperative staging: 256 threads, float4 copies ----
  {
    // A tile: 8192 floats -> 32 per thread (8 x float4)
    const int arow = threadIdx.x >> 4;            // 0..15
    const int ajb  = (threadIdx.x & 15) * 32;     // 32-float run
    const float* src = Adj + ((size_t)b * NN + i0 + arow) * NN + ajb;
    float* dst = Als + arow * NN + ajb;
    #pragma unroll
    for (int q = 0; q < 8; ++q)
      *reinterpret_cast<float4*>(dst + q * 4) =
          *reinterpret_cast<const float4*>(src + q * 4);
    // E2: 4096 floats -> 16 per thread (4 x float4)
    const int eh = threadIdx.x >> 5;              // 0..7
    const int ejb = (threadIdx.x & 31) * 16;
    const float* esrc = E2 + (size_t)(b * HH + eh) * NN + ejb;
    float* edst = E2ls + eh * NN + ejb;
    #pragma unroll
    for (int q = 0; q < 4; ++q)
      *reinterpret_cast<float4*>(edst + q * 4) =
          *reinterpret_cast<const float4*>(esrc + q * 4);
  }
  __syncthreads();

  const float e1 = E1[(size_t)(b * HH + h) * NN + i0 + l16];
  const float* Arow = Als + l16 * NN;
  const float* e2p  = E2ls + h * NN;

  // ---- pass 1: online softmax (masked) over this lane's j's ----
  float m = -3.0e38f, s = 0.0f;
  for (int j0 = 0; j0 < NN; j0 += 32) {
    float av[16], ev[16];
    ld16(Arow + j0 + base, av);
    ld16(e2p + j0 + base, ev);
    #pragma unroll
    for (int i = 0; i < 16; ++i) {
      if (av[i] > 0.5f) {
        float sc = e1 + ev[i];
        sc = fmaxf(sc, ALPHA * sc);       // leaky relu
        if (sc <= m) {                    // common path: one exp
          s += __expf(sc - m);
        } else {                          // new max: rescale
          s = s * __expf(m - sc) + 1.0f;
          m = sc;
        }
      }
    }
  }
  // merge partner lane (covers the other half of each j-step)
  const float mo = __shfl_xor(m, 16, 32);
  const float so = __shfl_xor(s, 16, 32);
  const float M = fmaxf(m, mo);
  const float S = s * __expf(m - M) + so * __expf(mo - M);

  // ---- pass 2: p = exp(score - M) into A-operand layout, WMMA vs Whh ----
  v8f c0 = {};
  v8f c1 = {};
  const bf16_t* b0p = WhT + (size_t)(b * EE + h * DD + l16) * NN;
  const bf16_t* b1p = WhT + (size_t)(b * EE + h * DD + 16 + l16) * NN;

  for (int j0 = 0; j0 < NN; j0 += 32) {
    // prefetch next j-step's B fragments (global_prefetch_b8)
    if (j0 + 32 < NN) {
      __builtin_prefetch(b0p + j0 + 32 + koff, 0, 3);
      __builtin_prefetch(b1p + j0 + 32 + koff, 0, 3);
    }
    float av[16], ev[16];
    ld16(Arow + j0 + base, av);
    ld16(e2p + j0 + base, ev);
    v16bf pmat;
    #pragma unroll
    for (int i = 0; i < 16; ++i) {
      float p = 0.0f;
      if (av[i] > 0.5f) {
        float sc = e1 + ev[i];
        sc = fmaxf(sc, ALPHA * sc);
        p = __expf(sc - M);
      }
      pmat[i] = (bf16_t)p;
    }
    // B fragments: WhT rows = feature cols, contiguous over j
    v8bf x0 = *reinterpret_cast<const v8bf*>(b0p + j0 + koff);
    v8bf x1 = *reinterpret_cast<const v8bf*>(b0p + j0 + koff + 16);
    v8bf y0 = *reinterpret_cast<const v8bf*>(b1p + j0 + koff);
    v8bf y1 = *reinterpret_cast<const v8bf*>(b1p + j0 + koff + 16);
    v16bf bm0, bm1;
    #pragma unroll
    for (int i = 0; i < 8; ++i) {
      bm0[i] = x0[i]; bm0[8 + i] = x1[i];
      bm1[i] = y0[i]; bm1[8 + i] = y1[i];
    }
    c0 = __builtin_amdgcn_wmma_f32_16x16x32_bf16(
        false, pmat, false, bm0, (short)0, c0, false, false);
    c1 = __builtin_amdgcn_wmma_f32_16x16x32_bf16(
        false, pmat, false, bm1, (short)0, c1, false, false);
  }

  // ---- 1/S per row (0 if no neighbors), ReLU, store [b,n,h*D+c] ----
  #pragma unroll
  for (int g = 0; g < 8; ++g) {
    const int rg = g + hi * 8;                    // row within tile
    const float sv = __shfl(S, rg, 32);           // lane rg holds row rg stats
    const float fac = (sv > 0.0f) ? (1.0f / sv) : 0.0f;
    const size_t o = ((size_t)b * NN + i0 + rg) * EE + h * DD + l16;
    out[o]      = fmaxf(c0[g] * fac, 0.0f);
    out[o + 16] = fmaxf(c1[g] * fac, 0.0f);
  }
}

// ---------------------------------------------------------------------------
extern "C" void kernel_launch(void* const* d_in, const int* in_sizes, int n_in,
                              void* d_out, int out_size, void* d_ws, size_t ws_size,
                              hipStream_t stream) {
  (void)in_sizes; (void)n_in; (void)out_size; (void)ws_size;
  const float* x  = (const float*)d_in[0];
  const float* A  = (const float*)d_in[1];
  const float* W  = (const float*)d_in[2];
  const float* bW = (const float*)d_in[3];
  const float* a  = (const float*)d_in[4];
  const float* ba = (const float*)d_in[5];
  float* out = (float*)d_out;

  // workspace layout (bytes, 256-aligned)
  char* w = (char*)d_ws;
  bf16_t* xb  = (bf16_t*)(w + 0);                       // 4 MB
  bf16_t* Wbt = (bf16_t*)(w + 4194304);                 // 128 KB
  bf16_t* WhT = (bf16_t*)(w + 4325376);                 // 4 MB
  float*  E1  = (float*) (w + 8519680);                 // 256 KB
  float*  E2  = (float*) (w + 8781824);                 // 256 KB

  const int prep_blocks = (BB * NN * EE + EE * EE + 255) / 256;  // 8448
  gat_prep <<<prep_blocks, 256, 0, stream>>>(x, W, xb, Wbt);
  gat_gemm1<<<(BB * NN) / 16, 256, 0, stream>>>(xb, Wbt, bW, WhT);        // 512
  gat_e12  <<<(2 * BB * HH * NN) / 256, 256, 0, stream>>>(WhT, a, ba, E1, E2); // 512
  gat_attn <<<BB * (NN / 16), 256, 0, stream>>>(A, E1, E2, WhT, out);     // 512
}
